// LogGaborConv2d_12841952215529
// MI455X (gfx1250) — matrix operand order repaired and checked by reading, not compiled, not run on hardware
//
#include <hip/hip_runtime.h>
#include <math.h>

typedef __attribute__((ext_vector_type(16))) __bf16 v16bf;
typedef __attribute__((ext_vector_type(8)))  float  v8f;

#define O_CH    128
#define I_CH    64
#define KTOT    576            // K order: k = (kh*3+kw)*64 + ci
#define KCHUNKS 18             // KTOT / 32
#define TILE_W  16
#define TILE_H  8
#define LDS_ROWS 10            // TILE_H + 2 (halo)
#define LDS_COLS 18            // TILE_W + 2 (halo)
#define CIS      72            // padded ci stride (bf16 elems): 16B-aligned frags,
                               // col stride = 36 dwords -> conflict-free b128
#define LDS_BF   (LDS_ROWS * LDS_COLS * CIS)   // 12960 ushort = 25920 B per array
#define SRC_TOTAL (I_CH * LDS_ROWS * LDS_COLS) // 11520 source fp32 elements
#define APACK_N  (8 * KCHUNKS * 32 * 16)       // 73728 bf16 per hi/lo array

// Split fp32 into bf16 hi (truncate) + bf16 lo (RNE of residual).
__device__ __forceinline__ void split_bf16(float v, unsigned short& h,
                                           unsigned short& l) {
    unsigned ub = __float_as_uint(v);
    unsigned hb = ub & 0xffff0000u;
    float lf = v - __uint_as_float(hb);          // exact residual
    unsigned lb = __float_as_uint(lf);
    lb += 0x7fffu + ((lb >> 16) & 1u);           // round-to-nearest-even
    h = (unsigned short)(ub >> 16);
    l = (unsigned short)(lb >> 16);
}

// Load 16 contiguous bf16 (32 B, 16B-aligned) as a WMMA fragment.
__device__ __forceinline__ v16bf load_bfrag(const unsigned short* p) {
    struct Q { uint4 a, b; } q;
    q.a = *(const uint4*)(p);
    q.b = *(const uint4*)(p + 8);
    return __builtin_bit_cast(v16bf, q);
}

// ---------------------------------------------------------------------------
// Phase 1: log-Gabor weights -> bf16 hi/lo A-fragments.
// Layout: Apack[o_tile(8)][kchunk(18)][lane(32)][slot j(16)], where slot j
// holds K = j + 8*((j>=8) + (lane>=16))  (ISA 16-bit A 16x32 layout),
// global k = kchunk*32 + K = khw*64 + ci.
// ---------------------------------------------------------------------------
__global__ void gabor_pack_weights(const float* __restrict__ freq,
                                   const float* __restrict__ theta,
                                   const float* __restrict__ sigma,
                                   const float* __restrict__ psi,
                                   const float* __restrict__ f0p,
                                   const float* __restrict__ th0p,
                                   const float* __restrict__ xg,
                                   const float* __restrict__ yg,
                                   unsigned short* __restrict__ ApackH,
                                   unsigned short* __restrict__ ApackL) {
    int id = blockIdx.x * blockDim.x + threadIdx.x;   // 0 .. 73727
    if (id >= APACK_N) return;

    int j    = id & 15;
    int lane = (id >> 4) & 31;
    int rest = id >> 9;                 // ot*18 + kc
    int kc   = rest % KCHUNKS;
    int ot   = rest / KCHUNKS;

    int K  = j + 8 * ((j >> 3) + (lane >> 4));      // 0..31
    int k  = kc * 32 + K;                           // 0..575
    int khw = k >> 6;                               // kh*3+kw, 0..8
    int ci  = k & 63;
    int kh  = khw / 3;
    int kw  = khw - kh * 3;
    int o   = ot * 16 + (lane & 15);

    int p = o * I_CH + ci;
    float th = theta[p];
    float sg = sigma[p];
    float fr = freq[p];
    float ps = psi[p];
    float f0 = f0p[0];
    float t0 = th0p[0];
    float x  = xg[khw];
    float y  = yg[khw];

    float c = __cosf(th), s = __sinf(th);
    float rotx =  x * c + y * s;
    float roty = -x * s + y * c;
    float r  = sqrtf(rotx * rotx + roty * roty + 0.001f);
    float lr = __logf(r) - __logf(f0);
    float dn = 2.0f * __logf(sg / f0);
    float q  = lr / dn;
    float g_rad = __expf(-q * q);
    float dth = th - t0;
    float g_ang = __expf(-(dth * dth) / (2.0f * sg * sg));
    float g = g_rad * g_ang * __cosf(fr * r + ps) /
              (2.0f * 3.14159265358979323846f * sg * sg);

    unsigned short h, l;
    split_bf16(g, h, l);
    ApackH[id] = h;
    ApackL[id] = l;
}

// ---------------------------------------------------------------------------
// Phase 2: implicit-GEMM conv, fp32-accuracy via bf16x3 emulation on
// V_WMMA_F32_16X16X32_BF16:  w*x ~= wh*xh + wh*xl + wl*xh.
// 256 threads = 8 waves; wave w owns output channels [16w,16w+16) for a
// 16(W) x 8(H) spatial tile. Input halo tile staged in LDS as bf16 hi/lo,
// [row][col][ci] with ci-stride 72 -> each B fragment is 2 aligned,
// bank-conflict-free ds_load_b128.
// ---------------------------------------------------------------------------
__global__ __launch_bounds__(256) void gabor_conv_wmma(
        const float* __restrict__ in,              // (8, 64, 256, 256)
        const unsigned short* __restrict__ ApackH,
        const unsigned short* __restrict__ ApackL,
        float* __restrict__ out) {                 // (8, 128, 256, 256)
    __shared__ unsigned short ldsH[LDS_BF];
    __shared__ unsigned short ldsL[LDS_BF];

    const int wt = blockIdx.x;            // 0..15  (W tiles)
    const int ht = blockIdx.y;            // 0..31  (H tiles)
    const int n  = blockIdx.z;            // 0..7   (batch)
    const int w0 = wt * TILE_W;
    const int h0 = ht * TILE_H;
    const int tid = threadIdx.x;

    // --- stage + split input tile (64 ci, rows h0-1..h0+8, cols w0-1..w0+16)
    for (int idx = tid; idx < SRC_TOTAL; idx += 256) {
        int ci  = idx / (LDS_ROWS * LDS_COLS);
        int rem = idx - ci * (LDS_ROWS * LDS_COLS);  // rem = row*18 + col
        int row = rem / LDS_COLS;
        int col = rem - row * LDS_COLS;
        int gh = h0 + row - 1;
        int gw = w0 + col - 1;
        float v = 0.0f;
        if ((unsigned)gh < 256u && (unsigned)gw < 256u)
            v = in[(((size_t)n * I_CH + ci) * 256 + gh) * 256 + gw];
        unsigned short h, l;
        split_bf16(v, h, l);
        int dst = rem * CIS + ci;
        ldsH[dst] = h;
        ldsL[dst] = l;
    }
    __syncthreads();

    const int wave  = tid >> 5;           // o_tile 0..7
    const int lane  = tid & 31;
    const int jcol  = lane & 15;          // N (spatial column in tile)
    const int khalf = lane >> 4;          // B half: K base 0 or 8

    // loop-invariant per-lane LDS element offset
    const int laneoff = jcol * CIS + khalf * 8;

    v8f acc[TILE_H];
#pragma unroll
    for (int t = 0; t < TILE_H; ++t) acc[t] = (v8f){};

    const unsigned short* __restrict__ AHw =
        ApackH + ((size_t)wave * KCHUNKS * 32 + lane) * 16;
    const unsigned short* __restrict__ ALw =
        ApackL + ((size_t)wave * KCHUNKS * 32 + lane) * 16;

    for (int kc = 0; kc < KCHUNKS; ++kc) {
        v16bf aH = load_bfrag(AHw + (size_t)kc * 32 * 16);  // 32B coalesced
        v16bf aL = load_bfrag(ALw + (size_t)kc * 32 * 16);

        // wave-uniform scalars (SALU)
        int khw   = kc >> 1;              // kh*3+kw
        int cbase = (kc & 1) << 5;        // ci base 0 or 32
        int kh    = khw / 3;
        int kw    = khw - kh * 3;
        int off   = (kh * LDS_COLS + kw) * CIS + cbase + laneoff;

#pragma unroll
        for (int t = 0; t < TILE_H; ++t) {
            int toff = off + t * (LDS_COLS * CIS);
            struct { uint4 a, b; } qH = { *(const uint4*)(ldsH + toff), *(const uint4*)(ldsH + toff + 16) };
            struct { uint4 a, b; } qL = { *(const uint4*)(ldsL + toff), *(const uint4*)(ldsL + toff + 16) };
            v16bf bH = __builtin_bit_cast(v16bf, qH);   // 2x ds_load_b128
            v16bf bL = __builtin_bit_cast(v16bf, qL);
            acc[t] = __builtin_amdgcn_wmma_f32_16x16x32_bf16(
                false, aH, false, bH, (short)0, acc[t], false, false);
            acc[t] = __builtin_amdgcn_wmma_f32_16x16x32_bf16(
                false, aH, false, bL, (short)0, acc[t], false, false);
            acc[t] = __builtin_amdgcn_wmma_f32_16x16x32_bf16(
                false, aL, false, bH, (short)0, acc[t], false, false);
        }
    }

    // --- store: C/D layout: VGPR v, lanes 0-15 -> M=v, lanes 16-31 -> M=v+8
    const int o_base = wave * 16 + khalf * 8;
#pragma unroll
    for (int t = 0; t < TILE_H; ++t) {
        int h = h0 + t;
        size_t base = (((size_t)n * O_CH + o_base) * 256 + h) * 256
                      + (size_t)(w0 + jcol);
#pragma unroll
        for (int v = 0; v < 8; ++v) {
            out[base + (size_t)v * 256 * 256] = acc[t][v];
        }
    }
}

extern "C" void kernel_launch(void* const* d_in, const int* in_sizes, int n_in,
                              void* d_out, int out_size, void* d_ws, size_t ws_size,
                              hipStream_t stream) {
    (void)in_sizes; (void)n_in; (void)out_size; (void)ws_size;

    const float* input = (const float*)d_in[0];
    const float* freq  = (const float*)d_in[1];
    const float* theta = (const float*)d_in[2];
    const float* sigma = (const float*)d_in[3];
    const float* psi   = (const float*)d_in[4];
    const float* f0    = (const float*)d_in[5];
    const float* th0   = (const float*)d_in[6];
    const float* xg    = (const float*)d_in[7];
    const float* yg    = (const float*)d_in[8];
    float* out = (float*)d_out;

    unsigned short* ApackH = (unsigned short*)d_ws;       // 73728 bf16
    unsigned short* ApackL = ApackH + APACK_N;            // 73728 bf16

    // Phase 1: 73728 packed weight slots (hi+lo)
    gabor_pack_weights<<<288, 256, 0, stream>>>(freq, theta, sigma, psi,
                                                f0, th0, xg, yg,
                                                ApackH, ApackL);

    // Phase 2: 16 x 32 x 8 workgroups of 256 threads
    dim3 grid(16, 32, 8);
    gabor_conv_wmma<<<grid, 256, 0, stream>>>(input, ApackH, ApackL, out);
}
